// ResonantRouter_43215960933249
// MI455X (gfx1250) — matrix operand order repaired and checked by reading, not compiled, not run on hardware
//
#include <hip/hip_runtime.h>
#include <hip/hip_bf16.h>
#include <math.h>
#include <stdint.h>

// ---------------------------------------------------------------------------
// ResonantRouter on MI455X (gfx1250, wave32, WMMA f16->f32, async-to-LDS)
// ---------------------------------------------------------------------------
#define D_DIM   128
#define HASHES  8
#define BUCKETS 1024
#define ROWS    8192          // 4 * 2048
#define STEPS   20
#define TM      64            // rows per workgroup in mega kernel
#define SW      136           // padded LDS row stride (halfs) for W / pattern chunks
#define SS      136           // padded LDS row stride (halfs) for state

#define USE_ASYNC_LDS 1       // gfx1250 GLOBAL_LOAD_ASYNC_TO_LDS_B128 path

typedef _Float16 v16h __attribute__((ext_vector_type(16)));
typedef _Float16 v8h  __attribute__((ext_vector_type(8)));
typedef float    v8f  __attribute__((ext_vector_type(8)));

// ---------------------------------------------------------------------------
// WMMA wrapper: D = A(16x32 f16) * B(32x16 f16) + C(16x16 f32)
// ---------------------------------------------------------------------------
__device__ inline v8f wmma_f16(v16h a, v16h b, v8f c) {
  return __builtin_amdgcn_wmma_f32_16x16x32_f16(
      /*neg_a=*/false, a, /*neg_b=*/false, b,
      /*c_mod=*/(short)0, c, /*reuse_a=*/false, /*reuse_b=*/false);
}

// A-fragment (16x32, f16) from LDS, row-major tile rows [rt*16 .. rt*16+15].
// Lane l (grp=l>>4, ln=l&15) holds row ln; K = (e<8?0:16) + 8*grp + (e&7).
__device__ inline v16h load_a_frag(const _Float16* s, int stride, int rt,
                                   int kb, int ln, int grp) {
  int base = (rt * 16 + ln) * stride + kb * 32 + grp * 8;
  v8h lo = *(const v8h*)(s + base);
  v8h hi = *(const v8h*)(s + base + 16);
  return __builtin_shufflevector(lo, hi, 0,1,2,3,4,5,6,7,8,9,10,11,12,13,14,15);
}

// B-fragment (32x16, f16) from LDS storing B TRANSPOSED as [n][k].
// Lane l holds column n = ln; K = (e<8?0:16) + 8*grp + (e&7).
__device__ inline v16h load_b_frag(const _Float16* s, int stride, int ct,
                                   int kb, int ln, int grp) {
  int base = (ct * 16 + ln) * stride + kb * 32 + grp * 8;
  v8h lo = *(const v8h*)(s + base);
  v8h hi = *(const v8h*)(s + base + 16);
  return __builtin_shufflevector(lo, hi, 0,1,2,3,4,5,6,7,8,9,10,11,12,13,14,15);
}

// ---------------------------------------------------------------------------
// Async global->LDS copy of one 128x128 f16 matrix into padded ([r][SW]) LDS.
// Each of 256 threads issues 8 GLOBAL_LOAD_ASYNC_TO_LDS_B128 (GVS form):
// 16B per lane per op. Tracked by ASYNCcnt (16 ops per wave per matrix pair).
// ---------------------------------------------------------------------------
__device__ inline void stage_matrix(const _Float16* __restrict__ gsrc,
                                    _Float16* ldst, int tid) {
#if USE_ASYNC_LDS
  uint64_t gb = (uint64_t)(uintptr_t)gsrc;
  uint32_t lb = (uint32_t)(uintptr_t)ldst;   // low 32b of LDS addr = LDS offset
  #pragma unroll
  for (int k = 0; k < 8; ++k) {
    int i = tid + k * 256;
    int r = i >> 4, cb = (i & 15) * 16;      // row, byte-column
    uint32_t go = (uint32_t)(r * (D_DIM * 2) + cb);
    uint32_t lo = lb + (uint32_t)(r * (SW * 2) + cb);
    asm volatile("global_load_async_to_lds_b128 %0, %1, %2"
                 :: "v"(lo), "v"(go), "s"(gb) : "memory");
  }
#else
  for (int i = tid; i < 128 * 16; i += 256) {
    int r = i >> 4, c = (i & 15) * 8;
    *(v8h*)(ldst + r * SW + c) = *(const v8h*)(gsrc + r * 128 + c);
  }
#endif
}

__device__ inline void wait_async_16() {
#if USE_ASYNC_LDS
  asm volatile("s_wait_asynccnt 0x10" ::: "memory");
#endif
}
__device__ inline void wait_async_0() {
#if USE_ASYNC_LDS
  asm volatile("s_wait_asynccnt 0x0" ::: "memory");
#endif
}

// ---------------------------------------------------------------------------
// Kernel 1: A = cos(clip(patterns)), B = sin(clip(patterns)) as f16
// ---------------------------------------------------------------------------
__global__ void rr_trig(const float* __restrict__ p,
                        _Float16* __restrict__ Ah,
                        _Float16* __restrict__ Bh, int n) {
  int i = blockIdx.x * 256 + threadIdx.x;
  if (i >= n) return;
  float v = p[i];
  v = fminf(fmaxf(v, -100.f), 100.f);
  Ah[i] = (_Float16)cosf(v);
  Bh[i] = (_Float16)sinf(v);
}

// ---------------------------------------------------------------------------
// Kernel 2: Wr = (A^T A + B^T B)/D, Wi = (A^T B - B^T A)/D, zero diag.
// Stored TRANSPOSED ([e][d]) as f16.  ~1 GFLOP: plain f32 VALU is fine.
// ---------------------------------------------------------------------------
__global__ void rr_wbuild(const _Float16* __restrict__ Ah,
                          const _Float16* __restrict__ Bh,
                          _Float16* __restrict__ WrT,
                          _Float16* __restrict__ WiT) {
  int h = blockIdx.x >> 7;     // 128 blocks per head (one per d)
  int d = blockIdx.x & 127;
  int e = threadIdx.x;         // 128 threads: one per e
  const _Float16* A = Ah + (size_t)h * BUCKETS * D_DIM;
  const _Float16* B = Bh + (size_t)h * BUCKETS * D_DIM;
  float wr = 0.f, wi = 0.f;
  for (int k = 0; k < BUCKETS; ++k) {
    float ad = (float)A[k * D_DIM + d];
    float bd = (float)B[k * D_DIM + d];
    float ae = (float)A[k * D_DIM + e];
    float be = (float)B[k * D_DIM + e];
    wr += ad * ae + bd * be;
    wi += ad * be - bd * ae;
  }
  wr *= (1.0f / 128.0f);
  wi *= (1.0f / 128.0f);
  if (d == e) { wr = 0.f; wi = 0.f; }
  size_t o = (size_t)h * D_DIM * D_DIM + (size_t)e * D_DIM + d; // transposed
  WrT[o] = (_Float16)wr;
  WiT[o] = (_Float16)wi;
}

// ---------------------------------------------------------------------------
// Kernel 3: mega kernel. One WG = (one head, 64 rows). State lives in LDS
// (double-buffered f16) for all 20 steps; fused similarity + argmax with
// double-buffered async pattern-chunk staging.
// 8 waves: wave w owns row-tile rt=w>>1 and col-tiles ct0=(w&1)*4 .. +3.
// ---------------------------------------------------------------------------
__global__ __launch_bounds__(256) void rr_mega(
    const float* __restrict__ tag,
    const _Float16* __restrict__ gWrT,
    const _Float16* __restrict__ gWiT,
    const _Float16* __restrict__ gA,
    const _Float16* __restrict__ gB,
    float* __restrict__ logits,
    float* __restrict__ idx_out) {
  extern __shared__ char smem[];
  _Float16* sWr  = (_Float16*)smem;                 // [128][SW] (chunk buf0: A)
  _Float16* sWi  = sWr + 128 * SW;                  // [128][SW] (chunk buf0: B)
  _Float16* sX   = sWi + 128 * SW;                  // [2][TM][SS]
  _Float16* sY   = sX + 2 * TM * SS;                // [2][TM][SS]
  _Float16* sC1  = sY + 2 * TM * SS;                // [2][128][SW] chunk buf1
  float*    redv = (float*)(sC1 + 2 * 128 * SW);    // [TM][32]
  int*      redi = (int*)(redv + TM * 32);          // [TM][32]

  const int tid  = threadIdx.x;
  const int h    = blockIdx.x & 7;
  const int rb   = blockIdx.x >> 3;
  const int row0 = rb * TM;
  const int lane = tid & 31;
  const int wave = tid >> 5;
  const int grp  = lane >> 4;
  const int ln   = lane & 15;
  const int rt   = wave >> 1;        // 0..3  (row tile)
  const int ct0  = (wave & 1) * 4;   // 0 or 4 (first col tile)

  // --- stage W (transposed f16) via async-to-LDS, overlap with init ------
  stage_matrix(gWrT + (size_t)h * D_DIM * D_DIM, sWr, tid);
  stage_matrix(gWiT + (size_t)h * D_DIM * D_DIM, sWi, tid);

  // --- init state: x = cos(tag), y = sin(tag) (nan->0, clip +-100) -------
  for (int i = tid; i < TM * 128; i += 256) {
    int r = i >> 7, c = i & 127;
    float v = tag[(size_t)(row0 + r) * 128 + c];
    v = (v == v) ? v : 0.f;                       // nan_to_num
    v = fminf(fmaxf(v, -100.f), 100.f);           // (+-inf clipped too)
    sX[r * SS + c] = (_Float16)cosf(v);
    sY[r * SS + c] = (_Float16)sinf(v);
  }
  wait_async_0();
#if !USE_ASYNC_LDS
  // fallback path already wrote via VGPRs; nothing extra
#endif
  __syncthreads();

  const float scale = 0.25f / 1024.0f;
  const float damp  = 0.98f;
  int cur = 0;

  // --- 20 resonance steps, state resident in LDS -------------------------
  for (int step = 0; step < STEPS; ++step) {
    _Float16* xc = sX + cur * (TM * SS);
    _Float16* yc = sY + cur * (TM * SS);
    _Float16* xn = sX + (cur ^ 1) * (TM * SS);
    _Float16* yn = sY + (cur ^ 1) * (TM * SS);

    v16h ax[4], ay[4];
    #pragma unroll
    for (int kb = 0; kb < 4; ++kb) {
      ax[kb] = load_a_frag(xc, SS, rt, kb, ln, grp);
      ay[kb] = load_a_frag(yc, SS, rt, kb, ln, grp);
    }

    #pragma unroll
    for (int t = 0; t < 4; ++t) {
      int ct = ct0 + t;
      v8f P = {}, Q = {}, R = {}, S = {};
      #pragma unroll
      for (int kb = 0; kb < 4; ++kb) {
        v16h bwr = load_b_frag(sWr, SW, ct, kb, ln, grp);
        v16h bwi = load_b_frag(sWi, SW, ct, kb, ln, grp);
        P = wmma_f16(ax[kb], bwr, P);   // x*Wr
        Q = wmma_f16(ay[kb], bwi, Q);   // y*Wi
        R = wmma_f16(ax[kb], bwi, R);   // x*Wi
        S = wmma_f16(ay[kb], bwr, S);   // y*Wr
      }
      // update + renormalize into the other buffer (c_r = P-Q, c_i = R+S)
      #pragma unroll
      for (int r = 0; r < 8; ++r) {
        int m = rt * 16 + grp * 8 + r;          // C layout: M = r + 8*grp
        int n = ct * 16 + ln;                   // N = lane&15
        float xv = (float)xc[m * SS + n];
        float yv = (float)yc[m * SS + n];
        float nx = (xv + scale * (P[r] - Q[r])) * damp;
        float ny = (yv + scale * (R[r] + S[r])) * damp;
        float inv = rsqrtf(nx * nx + ny * ny + 1e-12f);
        xn[m * SS + n] = (_Float16)(nx * inv);
        yn[m * SS + n] = (_Float16)(ny * inv);
      }
    }
    __syncthreads();
    cur ^= 1;
  }

  // --- fused similarity vs patterns + logits + argmax --------------------
  _Float16* xf = sX + cur * (TM * SS);
  _Float16* yf = sY + cur * (TM * SS);
  v16h ax[4], ay[4];
  #pragma unroll
  for (int kb = 0; kb < 4; ++kb) {
    ax[kb] = load_a_frag(xf, SS, rt, kb, ln, grp);
    ay[kb] = load_a_frag(yf, SS, rt, kb, ln, grp);
  }
  float bestv[8];
  int   besti[8];
  #pragma unroll
  for (int r = 0; r < 8; ++r) { bestv[r] = -1.f; besti[r] = 0; }

  // chunk buffers: buf0 reuses the W region (W reads finished above)
  _Float16* bufA[2] = { sWr, sC1 };
  _Float16* bufB[2] = { sWi, sC1 + 128 * SW };
  const _Float16* gAh = gA + (size_t)h * BUCKETS * D_DIM;
  const _Float16* gBh = gB + (size_t)h * BUCKETS * D_DIM;

  // prologue: prefetch chunks 0 and 1 (16 async ops per wave per chunk)
  stage_matrix(gAh + 0 * 128 * D_DIM, bufA[0], tid);
  stage_matrix(gBh + 0 * 128 * D_DIM, bufB[0], tid);
  stage_matrix(gAh + 1 * 128 * D_DIM, bufA[1], tid);
  stage_matrix(gBh + 1 * 128 * D_DIM, bufB[1], tid);

  for (int chunk = 0; chunk < BUCKETS / 128; ++chunk) {
    // retire this chunk's async loads (in-order completion)
    if (chunk < BUCKETS / 128 - 1) wait_async_16();
    else                           wait_async_0();
#if !USE_ASYNC_LDS
    {   // synchronous staging fallback
      const _Float16* ga = gAh + (size_t)chunk * 128 * D_DIM;
      const _Float16* gb = gBh + (size_t)chunk * 128 * D_DIM;
      _Float16* da = bufA[chunk & 1];
      _Float16* db = bufB[chunk & 1];
      for (int i = tid; i < 128 * 16; i += 256) {
        int r = i >> 4, c = (i & 15) * 8;
        *(v8h*)(da + r * SW + c) = *(const v8h*)(ga + r * 128 + c);
        *(v8h*)(db + r * SW + c) = *(const v8h*)(gb + r * 128 + c);
      }
    }
#endif
    __syncthreads();

    const _Float16* cA = bufA[chunk & 1];
    const _Float16* cB = bufB[chunk & 1];
    #pragma unroll
    for (int t = 0; t < 4; ++t) {
      int ct = ct0 + t;
      v8f dr = {}, d1 = {}, d2 = {};
      #pragma unroll
      for (int kb = 0; kb < 4; ++kb) {
        v16h bA = load_b_frag(cA, SW, ct, kb, ln, grp);
        v16h bB = load_b_frag(cB, SW, ct, kb, ln, grp);
        dr = wmma_f16(ax[kb], bA, dr);   // x.A
        dr = wmma_f16(ay[kb], bB, dr);   // + y.B
        d1 = wmma_f16(ay[kb], bA, d1);   // y.A
        d2 = wmma_f16(ax[kb], bB, d2);   // x.B  (di = d1 - d2)
      }
      #pragma unroll
      for (int r = 0; r < 8; ++r) {
        int m = rt * 16 + grp * 8 + r;
        int bucket = chunk * 128 + ct * 16 + ln;
        float rr = dr[r], ii = d1[r] - d2[r];
        float s = sqrtf(rr * rr + ii * ii + 1e-12f) * (1.0f / 128.0f);
        s = fminf(fmaxf(s, 0.f), 1.f);
        size_t grow = (size_t)(row0 + m);
        logits[(grow * HASHES + h) * BUCKETS + bucket] = s;
        if (s > bestv[r]) { bestv[r] = s; besti[r] = bucket; }
      }
    }
    __syncthreads();   // all reads of buf[chunk&1] done before re-staging it

    if (chunk + 2 < BUCKETS / 128) {
      stage_matrix(gAh + (size_t)(chunk + 2) * 128 * D_DIM, bufA[chunk & 1], tid);
      stage_matrix(gBh + (size_t)(chunk + 2) * 128 * D_DIM, bufB[chunk & 1], tid);
    }
  }

  // cross-lane / cross-wave argmax reduction (32 candidates per row)
  #pragma unroll
  for (int r = 0; r < 8; ++r) {
    int m = rt * 16 + grp * 8 + r;
    int slot = (wave & 1) * 16 + ln;
    redv[m * 32 + slot] = bestv[r];
    redi[m * 32 + slot] = besti[r];
  }
  __syncthreads();
  if (tid < TM) {
    float bv = -1.f; int bi = 0;
    for (int s2 = 0; s2 < 32; ++s2) {
      float v = redv[tid * 32 + s2];
      int   i2 = redi[tid * 32 + s2];
      if (v > bv || (v == bv && i2 < bi)) { bv = v; bi = i2; }
    }
    idx_out[(size_t)(row0 + tid) * HASHES + h] = (float)bi;
  }
}

// ---------------------------------------------------------------------------
extern "C" void kernel_launch(void* const* d_in, const int* in_sizes, int n_in,
                              void* d_out, int out_size, void* d_ws, size_t ws_size,
                              hipStream_t stream) {
  (void)in_sizes; (void)n_in; (void)out_size; (void)ws_size;
  const float* tag      = (const float*)d_in[0];   // (4,2048,128)
  const float* patterns = (const float*)d_in[1];   // (8,1024,128)

  char* ws = (char*)d_ws;
  size_t off = 0;
  const size_t nPat = (size_t)HASHES * BUCKETS * D_DIM;      // 1,048,576
  const size_t nW   = (size_t)HASHES * D_DIM * D_DIM;        // 131,072
  _Float16* Ah  = (_Float16*)(ws + off); off += nPat * 2;
  _Float16* Bh  = (_Float16*)(ws + off); off += nPat * 2;
  _Float16* WrT = (_Float16*)(ws + off); off += nW * 2;
  _Float16* WiT = (_Float16*)(ws + off); off += nW * 2;      // ~4.5 MB total

  float* logits = (float*)d_out;
  float* fidx   = logits + (size_t)ROWS * HASHES * BUCKETS;  // idx region

  rr_trig<<<dim3((unsigned)((nPat + 255) / 256)), dim3(256), 0, stream>>>(
      patterns, Ah, Bh, (int)nPat);

  rr_wbuild<<<dim3(HASHES * D_DIM), dim3(D_DIM), 0, stream>>>(
      Ah, Bh, WrT, WiT);

  const size_t smem = (size_t)2 * 128 * SW * 2      // W / chunk buf0
                    + (size_t)4 * TM * SS * 2       // x,y double-buffered
                    + (size_t)2 * 128 * SW * 2      // chunk buf1
                    + (size_t)TM * 32 * (4 + 4);    // argmax reduction
  rr_mega<<<dim3(HASHES * (ROWS / TM)), dim3(256), smem, stream>>>(
      tag, WrT, WiT, Ah, Bh, logits, fidx);
}
